// KBQADecoder_28166395527608
// MI455X (gfx1250) — matrix operand order repaired and checked by reading, not compiled, not run on hardware
//
#include <hip/hip_runtime.h>
#include <math.h>

// ---------------------------------------------------------------------------
// Types for CDNA5 WMMA (wave32): V_WMMA_F32_16X16X32_BF16
// ---------------------------------------------------------------------------
typedef __bf16 bf16_t;
typedef bf16_t v16bf __attribute__((ext_vector_type(16)));
typedef float  v8f   __attribute__((ext_vector_type(8)));

union FragU { uint4 q[2]; v16bf v; };

__device__ __forceinline__ unsigned short f2bf(float f) {
    unsigned int u = __float_as_uint(f);
    u += 0x7FFFu + ((u >> 16) & 1u);     // round-to-nearest-even
    return (unsigned short)(u >> 16);
}

__device__ __forceinline__ float gelu_erf(float v) {
    return 0.5f * v * (1.0f + erff(v * 0.70710678118654752f));
}

// ---------------------------------------------------------------------------
// Generic batched GEMM:  C[b] = alpha * (A[b] x W[b]^T) + bias  (+epilogue)
//   A : bf16 [M,K] row-major (lda), batch stride sA (elements)
//   W : bf16 [N,K] row-major (ldb), batch stride sW (0 => shared weights)
//   C : f32, normal store [M,N] (ldc) or transposed store [N,M] (flags&2)
//   flags: 1 = erf-GELU epilogue, 2 = transposed store
//
// Block: 256 threads = 8 waves; block tile 128(M) x 64(N); wave tile 16x64.
// The 64x32 bf16 weight strip is shared by all 8 waves, so it is staged
// through LDS (double-buffered, 80B padded row stride => conflict-free
// ds_load_b128 fragment reads), with the next strip's global load issued
// before the current strip's WMMAs to hide VMEM latency.
// ---------------------------------------------------------------------------
__global__ void k_gemm(const unsigned short* __restrict__ A, long long sA, int lda,
                       const unsigned short* __restrict__ W, long long sW, int ldb,
                       const float* __restrict__ bias, float alpha,
                       float* __restrict__ C, long long sC, int ldc,
                       int Mdim, int Ndim, int Kdim, int flags)
{
    // 2 buffers x 64 rows x 40 bf16 (80B padded stride; 32 bf16 payload)
    __shared__ __align__(16) unsigned short Bs[2][64 * 40];

    const int tid  = threadIdx.x;
    const int lane = tid & 31;
    const int wave = tid >> 5;
    const int m0   = blockIdx.y * 128 + wave * 16;
    const int n0   = blockIdx.x * 64;
    const int b    = blockIdx.z;

    const unsigned short* Ab = A + (size_t)b * sA;
    const unsigned short* Wb = W + (size_t)b * sW;

    const int half = lane >> 4;                 // 0 or 1
    int mrow = m0 + (lane & 15);                // A row for this lane
    if (mrow >= Mdim) mrow = Mdim - 1;          // clamp; C store is guarded

    // cooperative staging: thread -> (row 0..63, 16B chunk 0..3)
    const int srow = tid >> 2;
    const int schk = tid & 3;
    int wrow = n0 + srow;
    if (wrow >= Ndim) wrow = Ndim - 1;          // clamp (N always mult of 64 here)
    const unsigned short* wsrc = Wb + (size_t)wrow * ldb + (schk << 3);

    v8f acc[4] = {};

    uint4 stg = *(const uint4*)(wsrc);          // prologue: stage k-tile 0

    const int nsteps = Kdim >> 5;
    for (int s = 0; s < nsteps; ++s) {
        const int buf = s & 1;
        const int kb  = s << 5;

        // commit staged weight chunk to LDS
        *(uint4*)(&Bs[buf][srow * 40 + (schk << 3)]) = stg;
        __syncthreads();

        // issue next tile's global load early (overlaps ds reads + WMMA)
        if (s + 1 < nsteps) stg = *(const uint4*)(wsrc + kb + 32);

        // A fragment straight from global (rows are unique per wave)
        FragU fa;
        const unsigned short* pa = Ab + (size_t)mrow * lda + kb + (half << 3);
        fa.q[0] = *(const uint4*)(pa);
        fa.q[1] = *(const uint4*)(pa + 16);

#pragma unroll
        for (int t = 0; t < 4; ++t) {
            FragU fb;
            const unsigned short* bp =
                &Bs[buf][(16 * t + (lane & 15)) * 40 + (half << 3)];
            fb.q[0] = *(const uint4*)(bp);
            fb.q[1] = *(const uint4*)(bp + 16);
            acc[t] = __builtin_amdgcn_wmma_f32_16x16x32_bf16(
                false, fa.v, false, fb.v, (short)0, acc[t], false, false);
        }
        // single barrier per step is sufficient with double buffering:
        // a buffer is overwritten two steps later, past the next barrier.
    }

    // Epilogue / store.  C/D layout: VGPR r, lane L -> row m0 + r + 8*(L>=16),
    // col n0 + 16*t + (L&15).
    float* Cb = C + (size_t)b * sC;
    const int mb = m0 + (half << 3);
#pragma unroll
    for (int t = 0; t < 4; ++t) {
        const int n = n0 + 16 * t + (lane & 15);
        if (n >= Ndim) continue;
        const float bv = bias ? bias[n] : 0.0f;
#pragma unroll
        for (int r = 0; r < 8; ++r) {
            const int m = mb + r;
            if (m >= Mdim) continue;
            float v = alpha * acc[t][r] + bv;
            if (flags & 1) v = gelu_erf(v);
            if (flags & 2) Cb[(size_t)n * ldc + m] = v;
            else           Cb[(size_t)m * ldc + n] = v;
        }
    }
}

// ---------------------------------------------------------------------------
// fp32 -> bf16 conversion
// ---------------------------------------------------------------------------
__global__ void k_cvt(const float* __restrict__ src, unsigned short* __restrict__ dst,
                      long long n)
{
    long long i = (long long)blockIdx.x * blockDim.x + threadIdx.x;
    if (i < n) dst[i] = f2bf(src[i]);
}

// ---------------------------------------------------------------------------
// Build bf16 word_states [B, 3072, 512] = concat(fix, rel, type, var)
// ---------------------------------------------------------------------------
__global__ void k_build_word(unsigned short* __restrict__ wordb,
                             const float* __restrict__ fix,
                             const float* __restrict__ rel,
                             const float* __restrict__ ty,
                             const float* __restrict__ var)
{
    long long i = (long long)blockIdx.x * blockDim.x + threadIdx.x;
    const long long total = 8LL * 3072 * 512;
    if (i >= total) return;
    const int h = (int)(i & 511);
    const long long r = i >> 9;
    const int j = (int)(r % 3072);
    const int b = (int)(r / 3072);
    float v;
    if (j < 2048)      v = fix[(size_t)j * 512 + h];
    else if (j < 2560) v = rel[((size_t)b * 512 + (j - 2048)) * 512 + h];
    else if (j < 2816) v = ty [((size_t)b * 256 + (j - 2560)) * 512 + h];
    else               v = var[((size_t)b * 256 + (j - 2816)) * 512 + h];
    wordb[i] = f2bf(v);
}

// ---------------------------------------------------------------------------
// Gather embeddings: xcat[b*T+t, 0:512]=word[b,dec], [512:1024]=word[b,mix]
// ---------------------------------------------------------------------------
__global__ void k_gather(unsigned short* __restrict__ xcat,
                         const unsigned short* __restrict__ wordb,
                         const int* __restrict__ dec, const int* __restrict__ mix)
{
    long long i = (long long)blockIdx.x * blockDim.x + threadIdx.x;
    if (i >= 1024LL * 1024) return;
    const int c = (int)(i & 1023);
    const long long row = i >> 10;               // b*128+t
    const int b = (int)(row >> 7);
    const int id = (c < 512) ? dec[row] : mix[row];
    const int h = c & 511;
    xcat[i] = wordb[((size_t)b * 3072 + id) * 512 + h];
}

// ---------------------------------------------------------------------------
// GRU scan: one block per batch; xp = x@w_ih^T+b_ih precomputed (WMMA GEMM).
// ---------------------------------------------------------------------------
__global__ void k_gru(const float* __restrict__ xp,   // [B,T,1536]
                      const float* __restrict__ h0,   // [B,512]
                      const float* __restrict__ w_hh, // [1536,512]
                      const float* __restrict__ b_hh, // [1536]
                      float* __restrict__ inter,      // [B,T,512]
                      float* __restrict__ hT)         // [B,512]
{
    const int b = blockIdx.x;
    const int tid = threadIdx.x;
    __shared__ float h[512];
    __shared__ float hp[1536];
    for (int i = tid; i < 512; i += 256) h[i] = h0[b * 512 + i];
    __syncthreads();
    for (int t = 0; t < 128; ++t) {
        for (int j = tid; j < 1536; j += 256) {
            const float* wr = w_hh + (size_t)j * 512;
            float acc = b_hh[j];
            for (int k = 0; k < 512; ++k) acc = fmaf(h[k], wr[k], acc);
            hp[j] = acc;
        }
        __syncthreads();
        const float* xpt = xp + ((size_t)b * 128 + t) * 1536;
        float* out = inter + ((size_t)b * 128 + t) * 512;
        for (int i = tid; i < 512; i += 256) {
            const float r = 1.0f / (1.0f + __expf(-(xpt[i]        + hp[i])));
            const float z = 1.0f / (1.0f + __expf(-(xpt[512 + i]  + hp[512 + i])));
            const float n = tanhf(xpt[1024 + i] + r * hp[1024 + i]);
            const float hn = (1.0f - z) * n + z * h[i];
            out[i] = hn;
            h[i] = hn;
        }
        __syncthreads();
    }
    for (int i = tid; i < 512; i += 256) hT[b * 512 + i] = h[i];
}

// ---------------------------------------------------------------------------
// Row softmax with optional boolean mask; writes bf16 probabilities.
// One block (256 threads) per row.
// ---------------------------------------------------------------------------
__global__ void k_softmax(const float* __restrict__ s,
                          const unsigned char* __restrict__ mask,
                          unsigned short* __restrict__ p, int width)
{
    const int row = blockIdx.x;
    const int tid = threadIdx.x;
    const float* sr = s + (size_t)row * width;
    const unsigned char* mr = mask ? (mask + (size_t)row * width) : nullptr;
    __shared__ float red[256];

    float lmax = -3.0e38f;
    for (int i = tid; i < width; i += 256) {
        float v = sr[i];
        if (mr && !mr[i]) v = -3.0e38f;
        lmax = fmaxf(lmax, v);
    }
    red[tid] = lmax; __syncthreads();
    for (int o = 128; o > 0; o >>= 1) {
        if (tid < o) red[tid] = fmaxf(red[tid], red[tid + o]);
        __syncthreads();
    }
    const float smax = red[0]; __syncthreads();

    float lsum = 0.0f;
    for (int i = tid; i < width; i += 256) {
        float v = sr[i];
        if (mr && !mr[i]) v = -3.0e38f;
        lsum += __expf(v - smax);
    }
    red[tid] = lsum; __syncthreads();
    for (int o = 128; o > 0; o >>= 1) {
        if (tid < o) red[tid] += red[tid + o];
        __syncthreads();
    }
    const float inv = 1.0f / red[0];

    for (int i = tid; i < width; i += 256) {
        float v = sr[i];
        if (mr && !mr[i]) v = -3.0e38f;
        p[(size_t)row * width + i] = f2bf(__expf(v - smax) * inv);
    }
}

// ---------------------------------------------------------------------------
// LayerNorm (optional residual), writes fp32 and optional bf16 copy.
// One block (256 threads) per row; width = 512.
// ---------------------------------------------------------------------------
__global__ void k_layernorm(const float* __restrict__ x, const float* __restrict__ res,
                            const float* __restrict__ g, const float* __restrict__ bb,
                            float* __restrict__ yf, unsigned short* __restrict__ yb,
                            int width)
{
    const int row = blockIdx.x;
    const int tid = threadIdx.x;
    const float* xr = x + (size_t)row * width;
    const float* rr = res ? (res + (size_t)row * width) : nullptr;
    __shared__ float s1[256];
    __shared__ float s2[256];

    float lsum = 0.0f, lsq = 0.0f;
    for (int i = tid; i < width; i += 256) {
        float v = xr[i] + (rr ? rr[i] : 0.0f);
        lsum += v; lsq += v * v;
    }
    s1[tid] = lsum; s2[tid] = lsq; __syncthreads();
    for (int o = 128; o > 0; o >>= 1) {
        if (tid < o) { s1[tid] += s1[tid + o]; s2[tid] += s2[tid + o]; }
        __syncthreads();
    }
    const float mean = s1[0] / width;
    const float var  = s2[0] / width - mean * mean;
    const float rstd = rsqrtf(var + 1e-5f);

    for (int i = tid; i < width; i += 256) {
        float v = xr[i] + (rr ? rr[i] : 0.0f);
        float y = (v - mean) * rstd * g[i] + bb[i];
        yf[(size_t)row * width + i] = y;
        if (yb) yb[(size_t)row * width + i] = f2bf(y);
    }
}

// ---------------------------------------------------------------------------
// Apply variable-mask penalty on logit columns [2816, 3072)
// ---------------------------------------------------------------------------
__global__ void k_mask_logits(float* __restrict__ logits,
                              const unsigned char* __restrict__ vmask)
{
    int i = blockIdx.x * blockDim.x + threadIdx.x;      // over B*T*V = 262144
    if (i >= 8 * 128 * 256) return;
    if (!vmask[i]) {
        const int v = i & 255;
        const int row = i >> 8;                          // b*128+t
        logits[(size_t)row * 3072 + 2816 + v] += -3.402823466e38f;
    }
}

// ---------------------------------------------------------------------------
// Host orchestration
// ---------------------------------------------------------------------------
extern "C" void kernel_launch(void* const* d_in, const int* in_sizes, int n_in,
                              void* d_out, int out_size, void* d_ws, size_t ws_size,
                              hipStream_t stream)
{
    (void)in_sizes; (void)n_in; (void)out_size; (void)ws_size;
    constexpr int Bc = 8, Tc = 128, Hc = 512, Mc = 512, Vc = 256;
    constexpr int Wtot = 3072, BT = Bc * Tc;
    constexpr long long THW = (long long)Tc * Hc;

    // ---- inputs (setup_inputs dict order, params flattened) ----
    const int*           dec   = (const int*)d_in[0];
    const int*           mix   = (const int*)d_in[1];
    const float*         h0    = (const float*)d_in[2];
    const float*         rel   = (const float*)d_in[3];
    const float*         typ   = (const float*)d_in[4];
    const float*         var   = (const float*)d_in[5];
    const unsigned char* vmask = (const unsigned char*)d_in[6];
    const float*         memst = (const float*)d_in[7];
    const float*         fix   = (const float*)d_in[8];
    const float*         Wmix  = (const float*)d_in[9];
    const float*         bmix  = (const float*)d_in[10];
    const float*         w_ih  = (const float*)d_in[11];
    const float*         w_hh  = (const float*)d_in[12];
    const float*         b_ih  = (const float*)d_in[13];
    const float*         b_hh  = (const float*)d_in[14];
    // attn params: [15..30]=mem, [31..46]=var : Wq,Wk,Wv,Wd,W0,W1, bq,bk,bv,bd,b0,b1w, g1,be1,g2,be2
    const float* Wt  = (const float*)d_in[47];
    const float* bt  = (const float*)d_in[48];
    const float* gt  = (const float*)d_in[49];
    const float* bet = (const float*)d_in[50];

    float* logits = (float*)d_out;                          // [8,128,3072]
    float* hT_out = (float*)d_out + (size_t)Bc * Tc * Wtot; // [8,512]

    // ---- workspace bump allocator ----
    char* wsb = (char*)d_ws; size_t off = 0;
    auto alloc = [&](size_t bytes) -> void* {
        void* p = wsb + off; off = (off + bytes + 255) & ~(size_t)255; return p;
    };
    auto cvt = [&](const float* s, unsigned short* d, long long n) {
        k_cvt<<<(unsigned)((n + 255) / 256), 256, 0, stream>>>(s, d, n);
    };
    auto gemm = [&](const unsigned short* A, long long sA, int lda,
                    const unsigned short* W, long long sW, int ldb,
                    const float* bias, float alpha,
                    float* C, long long sC, int ldc,
                    int M_, int N_, int K_, int flags, int batch) {
        dim3 g((N_ + 63) / 64, (M_ + 127) / 128, batch);
        k_gemm<<<g, 256, 0, stream>>>(A, sA, lda, W, sW, ldb, bias, alpha,
                                      C, sC, ldc, M_, N_, K_, flags);
    };

    // buffers
    unsigned short* wordb = (unsigned short*)alloc((size_t)Bc * Wtot * Hc * 2);
    unsigned short* Wmixb = (unsigned short*)alloc((size_t)512 * 1024 * 2);
    unsigned short* wihb  = (unsigned short*)alloc((size_t)1536 * 512 * 2);
    unsigned short* Wtb   = (unsigned short*)alloc((size_t)512 * 512 * 2);
    unsigned short* attnWb[2][6];
    const float*    attnP[2][16];
    for (int s = 0; s < 2; ++s) {
        for (int j = 0; j < 16; ++j) attnP[s][j] = (const float*)d_in[15 + 16 * s + j];
        for (int j = 0; j < 6;  ++j) attnWb[s][j] = (unsigned short*)alloc((size_t)512 * 512 * 2);
    }
    unsigned short* xcatb = (unsigned short*)alloc((size_t)BT * 1024 * 2);
    float*          xf    = (float*)alloc((size_t)BT * Hc * 4);
    unsigned short* xb    = (unsigned short*)alloc((size_t)BT * Hc * 2);
    float*          xp    = (float*)alloc((size_t)BT * 1536 * 4);
    float*          interf = (float*)alloc((size_t)BT * Hc * 4);
    unsigned short* interb = (unsigned short*)alloc((size_t)BT * Hc * 2);
    unsigned short* membuf = (unsigned short*)alloc((size_t)Bc * Mc * Hc * 2);
    // attention scratch (sized for Mkv = 512)
    float*          qf  = (float*)alloc((size_t)BT * Hc * 4);
    unsigned short* qb  = (unsigned short*)alloc((size_t)BT * Hc * 2);
    float*          kf  = (float*)alloc((size_t)Bc * Mc * Hc * 4);
    unsigned short* kb  = (unsigned short*)alloc((size_t)Bc * Mc * Hc * 2);
    float*          vTf = (float*)alloc((size_t)Bc * Hc * Mc * 4);
    unsigned short* vTb = (unsigned short*)alloc((size_t)Bc * Hc * Mc * 2);
    float*          sf  = (float*)alloc((size_t)BT * Mc * 4);
    unsigned short* pb  = (unsigned short*)alloc((size_t)BT * Mc * 2);
    float*          aof = (float*)alloc((size_t)BT * Hc * 4);
    unsigned short* aob = (unsigned short*)alloc((size_t)BT * Hc * 2);
    float*          prjf = (float*)alloc((size_t)BT * Hc * 4);
    float*          ln1f = (float*)alloc((size_t)BT * Hc * 4);
    unsigned short* ln1b = (unsigned short*)alloc((size_t)BT * Hc * 2);
    float*          ffnf = (float*)alloc((size_t)BT * Hc * 4);
    unsigned short* ffnb = (unsigned short*)alloc((size_t)BT * Hc * 2);
    float*          out2f = (float*)alloc((size_t)BT * Hc * 4);
    float*          o1f = (float*)alloc((size_t)BT * Hc * 4);
    unsigned short* o1b = (unsigned short*)alloc((size_t)BT * Hc * 2);
    float*          o2f = (float*)alloc((size_t)BT * Hc * 4);
    unsigned short* o2b = (unsigned short*)alloc((size_t)BT * Hc * 2);
    float*          trg = (float*)alloc((size_t)BT * Hc * 4);
    float*          trf = (float*)alloc((size_t)BT * Hc * 4);
    unsigned short* trb = (unsigned short*)alloc((size_t)BT * Hc * 2);

    // ---- 1) word states + weight conversion ----
    {
        const long long n = (long long)Bc * Wtot * Hc;
        k_build_word<<<(unsigned)((n + 255) / 256), 256, 0, stream>>>(wordb, fix, rel, typ, var);
    }
    cvt(Wmix, Wmixb, 512LL * 1024);
    cvt(w_ih, wihb, 1536LL * 512);
    cvt(Wt,   Wtb,  512LL * 512);
    for (int s = 0; s < 2; ++s)
        for (int j = 0; j < 6; ++j)
            cvt(attnP[s][j], attnWb[s][j], 512LL * 512);

    // ---- 2) embedding gather + mix linear ----
    k_gather<<<(unsigned)((1024LL * 1024 + 255) / 256), 256, 0, stream>>>(xcatb, wordb, dec, mix);
    gemm(xcatb, 0, 1024, Wmixb, 0, 1024, bmix, 1.0f, xf, 0, 512, BT, 512, 1024, 0, 1);
    cvt(xf, xb, (long long)BT * Hc);

    // ---- 3) GRU: xp = x @ w_ih^T + b_ih (WMMA), then sequential scan ----
    gemm(xb, 0, 512, wihb, 0, 512, b_ih, 1.0f, xp, 0, 1536, BT, 1536, 512, 0, 1);
    k_gru<<<Bc, 256, 0, stream>>>(xp, h0, w_hh, b_hh, interf, hT_out);
    cvt(interf, interb, (long long)BT * Hc);

    // ---- 4) attention blocks ----
    cvt(memst, membuf, (long long)Bc * Mc * Hc);
    const float inv_sqrt_h = 0.044194173824159216f;   // 1/sqrt(512)

    auto run_attn = [&](const float* xin_f, const unsigned short* xin_b,
                        const unsigned short* kvb, long long kvStride, int Mkv,
                        const unsigned char* mask, int s,
                        float* outf, unsigned short* outb) {
        const float** P = attnP[s];
        unsigned short** W = attnWb[s];
        // projections
        gemm(xin_b, THW, Hc, W[0], 0, Hc, P[6 + 0], 1.0f, qf, THW, Hc, Tc, Hc, Hc, 0, Bc);
        gemm(kvb, kvStride, Hc, W[1], 0, Hc, P[6 + 1], 1.0f, kf, (long long)Mkv * Hc, Hc, Mkv, Hc, Hc, 0, Bc);
        gemm(kvb, kvStride, Hc, W[2], 0, Hc, P[6 + 2], 1.0f, vTf, (long long)Hc * Mkv, Mkv, Mkv, Hc, Hc, 2, Bc);
        cvt(qf, qb, (long long)BT * Hc);
        cvt(kf, kb, (long long)Bc * Mkv * Hc);
        cvt(vTf, vTb, (long long)Bc * Hc * Mkv);
        // scores + softmax
        gemm(qb, THW, Hc, kb, (long long)Mkv * Hc, Hc, nullptr, inv_sqrt_h,
             sf, (long long)Tc * Mkv, Mkv, Tc, Mkv, Hc, 0, Bc);
        k_softmax<<<BT, 256, 0, stream>>>(sf, mask, pb, Mkv);
        // attention output
        gemm(pb, (long long)Tc * Mkv, Mkv, vTb, (long long)Hc * Mkv, Mkv, nullptr, 1.0f,
             aof, THW, Hc, Tc, Hc, Mkv, 0, Bc);
        cvt(aof, aob, (long long)BT * Hc);
        gemm(aob, THW, Hc, W[3], 0, Hc, P[6 + 3], 1.0f, prjf, THW, Hc, Tc, Hc, Hc, 0, Bc);
        k_layernorm<<<BT, 256, 0, stream>>>(prjf, xin_f, P[12], P[13], ln1f, ln1b, Hc);
        // FFN (erf-GELU fused in GEMM epilogue)
        gemm(ln1b, THW, Hc, W[4], 0, Hc, P[6 + 4], 1.0f, ffnf, THW, Hc, Tc, Hc, Hc, 1, Bc);
        cvt(ffnf, ffnb, (long long)BT * Hc);
        gemm(ffnb, THW, Hc, W[5], 0, Hc, P[6 + 5], 1.0f, out2f, THW, Hc, Tc, Hc, Hc, 0, Bc);
        k_layernorm<<<BT, 256, 0, stream>>>(out2f, ln1f, P[14], P[15], outf, outb, Hc);
    };

    run_attn(interf, interb, membuf, (long long)Mc * Hc, Mc, nullptr, 0, o1f, o1b);
    // variable states are already bf16 inside wordb at row offset 2816
    run_attn(o1f, o1b, wordb + (size_t)2816 * Hc, (long long)Wtot * Hc, Vc, vmask, 1, o2f, o2b);

    // ---- 5) transform: Linear -> GELU -> LayerNorm ----
    gemm(o2b, THW, Hc, Wtb, 0, Hc, bt, 1.0f, trg, THW, Hc, Tc, Hc, Hc, 1, Bc);
    k_layernorm<<<BT, 256, 0, stream>>>(trg, nullptr, gt, bet, trf, trb, Hc);

    // ---- 6) logits = tr @ word_states^T  (+ variable mask) ----
    gemm(trb, THW, Hc, wordb, (long long)Wtot * Hc, Hc, nullptr, 1.0f,
         logits, (long long)Tc * Wtot, Wtot, Tc, Wtot, Hc, 0, Bc);
    k_mask_logits<<<(Bc * Tc * Vc + 255) / 256, 256, 0, stream>>>(logits, vmask);
}
